// MultiTaskGNN_25967372272018
// MI455X (gfx1250) — matrix operand order repaired and checked by reading, not compiled, not run on hardware
//
#include <hip/hip_runtime.h>
#include <hip/hip_bf16.h>

// MultiTaskGNN forward for gfx1250 (MI455X).
// Key algebraic fusion: msg = relu( (h@(node_w@msg_w_top)+b_eff)[src] + edge_attr@(edge_w@msg_w_bot) )
// -> dense GEMMs become N x 128 x 128 bf16 WMMA jobs; edge pass is gather + relu + atomic scatter.

typedef __attribute__((ext_vector_type(16))) __bf16 v16bf;
typedef __attribute__((ext_vector_type(8)))  float  v8f;

constexpr int Nn = 50000, Ee = 800000, Gg = 50;
constexpr int ND = 8, ED = 2, Hh = 128, Ll = 3;
constexpr int PACK_DW = (Hh * Hh) / 2; // packed bf16 dwords per 128x128 matrix

__device__ __forceinline__ unsigned short f2bf(float f) {
  union { float f; unsigned u; } v; v.f = f;
  unsigned r = v.u + 0x7FFFu + ((v.u >> 16) & 1u); // RNE
  return (unsigned short)(r >> 16);
}

union Frag { v16bf v; uint4 u[2]; };

// ---------------------------------------------------------------------------
// WMMA GEMM: C[nrows x 128] = A(bf16 row-major) @ Bpack + bias, optional relu.
// Bpack layout: [nt(8)][kt(4)][lane(32)][r(8)] dwords (2 packed bf16 each).
// One wave per 16-row tile; EXEC stays all-ones (wave-uniform guard only).
// ---------------------------------------------------------------------------
__global__ __launch_bounds__(256)
void gemm_bf16_wmma(const unsigned short* __restrict__ A,
                    const unsigned int* __restrict__ Bpack,
                    const float* __restrict__ bias,
                    float* __restrict__ C,
                    int nrows, int do_relu)
{
  const int wave = threadIdx.x >> 5;
  const int lane = threadIdx.x & 31;
  const int mt = blockIdx.x * 8 + wave;
  if (mt * 16 >= nrows) return;                 // wave-uniform
  const int g  = lane >> 4;
  const int ln = lane & 15;

  // A fragments: 16-bit A 16x32 layout => lanes 0-15 hold K in {8g..8g+7, 8g+16..8g+23}
  const unsigned short* Arow = A + (size_t)(mt * 16 + ln) * Hh;
  Frag a[4];
#pragma unroll
  for (int kt = 0; kt < 4; ++kt) {
    int k0 = kt * 32 + 8 * g;
    a[kt].u[0] = *(const uint4*)(Arow + k0);
    a[kt].u[1] = *(const uint4*)(Arow + k0 + 16);
  }

#pragma unroll
  for (int nt = 0; nt < 8; ++nt) {
    v8f acc = {0.f, 0.f, 0.f, 0.f, 0.f, 0.f, 0.f, 0.f};
#pragma unroll
    for (int kt = 0; kt < 4; ++kt) {
      Frag b;
      const uint4* bp = (const uint4*)(Bpack + (((nt * 4 + kt) * 32 + lane) * 8));
      b.u[0] = bp[0];
      b.u[1] = bp[1];
      acc = __builtin_amdgcn_wmma_f32_16x16x32_bf16(
          false, a[kt].v, false, b.v, (short)0, acc, false, false);
    }
    const float bv = bias[nt * 16 + ln];        // column N = nt*16 + (lane&15)
    float* Cbase = C + (size_t)(mt * 16) * Hh + nt * 16 + ln;
#pragma unroll
    for (int v = 0; v < 8; ++v) {               // row M = v + 8g
      float o = acc[v] + bv;
      if (do_relu) o = fmaxf(o, 0.f);
      Cbase[(size_t)(v + 8 * g) * Hh] = o;
    }
  }
}

// ---------------------------------------------------------------------------
// Precompute: Weff[l] = node_w[l] @ msg_w_top[l]   (H x H per layer)
// ---------------------------------------------------------------------------
__global__ __launch_bounds__(256)
void compute_weff(const float* __restrict__ node_w, const float* __restrict__ msg_w,
                  float* __restrict__ Weff)
{
  int idx = blockIdx.x * 256 + threadIdx.x;
  if (idx >= Ll * Hh * Hh) return;
  int l = idx / (Hh * Hh);
  int kn = idx - l * Hh * Hh;
  int k = kn >> 7, n = kn & 127;
  const float* nr = node_w + (size_t)(l * Hh + k) * Hh;
  const float* mw = msg_w + (size_t)l * 2 * Hh * Hh + n;
  float s = 0.f;
  for (int j = 0; j < Hh; ++j) s += nr[j] * mw[j * Hh];
  Weff[idx] = s;
}

// Pack a 128x128 f32 row-major matrix into the WMMA-B bf16 fragment layout.
__global__ __launch_bounds__(256)
void pack_b(const float* __restrict__ W, unsigned int* __restrict__ P)
{
  int idx = blockIdx.x * 256 + threadIdx.x;
  if (idx >= PACK_DW) return;
  int r    = idx & 7;
  int lane = (idx >> 3) & 31;
  int kt   = (idx >> 8) & 3;
  int nt   = idx >> 10;
  int g = lane >> 4;
  int n = nt * 16 + (lane & 15);
  int k = kt * 32 + 8 * g + 2 * r + ((r >> 2) << 3);   // B: VGPR r, K=8g+2r (+8 for r>=4), pairs K,K+1
  unsigned lo = f2bf(W[k * Hh + n]);
  unsigned hi = f2bf(W[(k + 1) * Hh + n]);
  P[idx] = lo | (hi << 16);
}

// W2eff[l] = edge_w[l] @ msg_w_bot[l] (2 x H);  beff[l] = node_b@top + edge_b@bot + msg_b
__global__ __launch_bounds__(256)
void prep_edge_const(const float* __restrict__ node_b, const float* __restrict__ edge_w,
                     const float* __restrict__ edge_b, const float* __restrict__ msg_w,
                     const float* __restrict__ msg_b,
                     float* __restrict__ W2eff, float* __restrict__ beff)
{
  int idx = blockIdx.x * 256 + threadIdx.x;
  if (idx >= Ll * Hh) return;
  int l = idx / Hh, n = idx & 127;
  const float* mwl = msg_w + (size_t)l * 2 * Hh * Hh;
  float w0 = 0.f, w1 = 0.f, bb = msg_b[l * Hh + n];
  for (int j = 0; j < Hh; ++j) {
    float mtop = mwl[j * Hh + n];
    float mbot = mwl[(Hh + j) * Hh + n];
    w0 += edge_w[(l * ED + 0) * Hh + j] * mbot;
    w1 += edge_w[(l * ED + 1) * Hh + j] * mbot;
    bb += node_b[l * Hh + j] * mtop + edge_b[l * Hh + j] * mbot;
  }
  W2eff[(l * 2 + 0) * Hh + n] = w0;
  W2eff[(l * 2 + 1) * Hh + n] = w1;
  beff[l * Hh + n] = bb;
}

// h = x @ in_w + in_b ; also write bf16 copy for WMMA
__global__ __launch_bounds__(256)
void input_proj(const float* __restrict__ x, const float* __restrict__ w,
                const float* __restrict__ b, float* __restrict__ h,
                unsigned short* __restrict__ hb)
{
  int idx = blockIdx.x * 256 + threadIdx.x;
  if (idx >= Nn * Hh) return;
  int i = idx >> 7, j = idx & 127;
  float s = b[j];
#pragma unroll
  for (int d = 0; d < ND; ++d) s += x[i * ND + d] * w[d * Hh + j];
  h[idx] = s;
  hb[idx] = f2bf(s);
}

__global__ __launch_bounds__(256)
void zero_f4(float4* __restrict__ p, int n4)
{
  int i = blockIdx.x * 256 + threadIdx.x;
  if (i < n4) p[i] = make_float4(0.f, 0.f, 0.f, 0.f);
}

// Edge pass: 32 lanes per edge; msg = relu(hn2[src] + edge_attr@W2eff); atomic scatter to agg[dst].
__global__ __launch_bounds__(256)
void edge_pass(const float* __restrict__ hn2, const float* __restrict__ eattr,
               const int* __restrict__ ei, const float* __restrict__ W2,
               float* __restrict__ agg, int ne)
{
  int t = blockIdx.x * 256 + threadIdx.x;
  int e = t >> 5;
  if (e >= ne) return;
  int lane = t & 31;
  int src = ei[e];
  int dst = ei[ne + e];
  float a0 = eattr[2 * e], a1 = eattr[2 * e + 1];
  int j = lane * 4;
  float4 hv = *(const float4*)(hn2 + (size_t)src * Hh + j);
  float4 w0 = *(const float4*)(W2 + j);
  float4 w1 = *(const float4*)(W2 + Hh + j);
  float m0 = fmaxf(hv.x + a0 * w0.x + a1 * w1.x, 0.f);
  float m1 = fmaxf(hv.y + a0 * w0.y + a1 * w1.y, 0.f);
  float m2 = fmaxf(hv.z + a0 * w0.z + a1 * w1.z, 0.f);
  float m3 = fmaxf(hv.w + a0 * w0.w + a1 * w1.w, 0.f);
  float* ap = agg + (size_t)dst * Hh + j;
  unsafeAtomicAdd(ap + 0, m0);
  unsafeAtomicAdd(ap + 1, m1);
  unsafeAtomicAdd(ap + 2, m2);
  unsafeAtomicAdd(ap + 3, m3);
}

// LayerNorm(h + agg) * g + b ; one wave per node; writes f32 h_out and bf16 hb.
__global__ __launch_bounds__(256)
void layernorm_k(const float* __restrict__ h_in, const float* __restrict__ agg,
                 const float* __restrict__ g, const float* __restrict__ b,
                 float* __restrict__ h_out, unsigned short* __restrict__ hb, int n)
{
  int t = blockIdx.x * 256 + threadIdx.x;
  int i = t >> 5;
  if (i >= n) return;
  int lane = t & 31;
  int off = lane * 4;
  float4 hv = *(const float4*)(h_in + (size_t)i * Hh + off);
  float4 av = *(const float4*)(agg + (size_t)i * Hh + off);
  float v0 = hv.x + av.x, v1 = hv.y + av.y, v2 = hv.z + av.z, v3 = hv.w + av.w;
  float s = v0 + v1 + v2 + v3;
#pragma unroll
  for (int o = 16; o; o >>= 1) s += __shfl_xor(s, o, 32);
  float mu = s * (1.f / Hh);
  float d0 = v0 - mu, d1 = v1 - mu, d2 = v2 - mu, d3 = v3 - mu;
  float q = d0 * d0 + d1 * d1 + d2 * d2 + d3 * d3;
#pragma unroll
  for (int o = 16; o; o >>= 1) q += __shfl_xor(q, o, 32);
  float rs = rsqrtf(q * (1.f / Hh) + 1e-5f);
  float4 gv = *(const float4*)(g + off);
  float4 bv = *(const float4*)(b + off);
  float r0 = d0 * rs * gv.x + bv.x;
  float r1 = d1 * rs * gv.y + bv.y;
  float r2 = d2 * rs * gv.z + bv.z;
  float r3 = d3 * rs * gv.w + bv.w;
  *(float4*)(h_out + (size_t)i * Hh + off) = make_float4(r0, r1, r2, r3);
  uint2 pk;
  pk.x = (unsigned)f2bf(r0) | ((unsigned)f2bf(r1) << 16);
  pk.y = (unsigned)f2bf(r2) | ((unsigned)f2bf(r3) << 16);
  *(uint2*)(hb + (size_t)i * Hh + off) = pk;
}

// Second-stage heads: pf (H->3) and gen (H->2); one wave per node.
__global__ __launch_bounds__(256)
void head_pf_gen(const float* __restrict__ t1, const float* __restrict__ g1,
                 const float* __restrict__ pfw2, const float* __restrict__ pfb2,
                 const float* __restrict__ genw2, const float* __restrict__ genb2,
                 const float* __restrict__ gmask,
                 float* __restrict__ o_vmag, float* __restrict__ o_sin,
                 float* __restrict__ o_cos, float* __restrict__ o_gen, int n)
{
  int t = blockIdx.x * 256 + threadIdx.x;
  int i = t >> 5;
  if (i >= n) return;
  int lane = t & 31;
  const float* tr = t1 + (size_t)i * Hh + lane * 4;
  const float* gr = g1 + (size_t)i * Hh + lane * 4;
  float s0 = 0, s1 = 0, s2 = 0, q0 = 0, q1 = 0;
#pragma unroll
  for (int q = 0; q < 4; ++q) {
    int j = lane * 4 + q;
    float tv = tr[q], gv = gr[q];
    s0 += tv * pfw2[j * 3 + 0];
    s1 += tv * pfw2[j * 3 + 1];
    s2 += tv * pfw2[j * 3 + 2];
    q0 += gv * genw2[j * 2 + 0];
    q1 += gv * genw2[j * 2 + 1];
  }
#pragma unroll
  for (int o = 16; o; o >>= 1) {
    s0 += __shfl_xor(s0, o, 32);
    s1 += __shfl_xor(s1, o, 32);
    s2 += __shfl_xor(s2, o, 32);
    q0 += __shfl_xor(q0, o, 32);
    q1 += __shfl_xor(q1, o, 32);
  }
  if (lane == 0) {
    float o0 = s0 + pfb2[0], o1 = s1 + pfb2[1], o2 = s2 + pfb2[2];
    o_vmag[i] = (o0 > 30.f) ? o0 : log1pf(__expf(o0)); // softplus
    float nrm = sqrtf(o1 * o1 + o2 * o2 + 1e-8f);
    o_sin[i] = o1 / nrm;
    o_cos[i] = o2 / nrm;
    float m = gmask[i];
    o_gen[2 * i + 0] = (q0 + genb2[0]) * m;
    o_gen[2 * i + 1] = (q1 + genb2[1]) * m;
  }
}

// Per-graph mean + cost head (batch is contiguous: 1000 nodes per graph).
__global__ __launch_bounds__(128)
void graph_cost(const float* __restrict__ h, const float* __restrict__ w1,
                const float* __restrict__ b1, const float* __restrict__ w2,
                const float* __restrict__ b2, float* __restrict__ o_cost, int per_g)
{
  __shared__ float gm[Hh];
  __shared__ float hid[64];
  int g = blockIdx.x, j = threadIdx.x;
  const float* base = h + (size_t)g * per_g * Hh;
  float s = 0.f;
  for (int r = 0; r < per_g; ++r) s += base[(size_t)r * Hh + j];
  gm[j] = s / (float)per_g;
  __syncthreads();
  if (j < 64) {
    float v = b1[j];
    for (int k = 0; k < Hh; ++k) v += gm[k] * w1[k * 64 + j];
    hid[j] = fmaxf(v, 0.f);
  }
  __syncthreads();
  if (j == 0) {
    float v = b2[0];
    for (int k = 0; k < 64; ++k) v += hid[k] * w2[k];
    o_cost[g] = v;
  }
}

// Attention scores: a[i] = tanh(h@att_w1 + b1) @ att_w2 + b2; one wave per node.
__global__ __launch_bounds__(256)
void att_score(const float* __restrict__ h, const float* __restrict__ w1,
               const float* __restrict__ b1, const float* __restrict__ w2,
               const float* __restrict__ b2, float* __restrict__ a, int n)
{
  int t = blockIdx.x * 256 + threadIdx.x;
  int i = t >> 5;
  if (i >= n) return;
  int lane = t & 31;
  const float* hr = h + (size_t)i * Hh;
  float acc = 0.f;
#pragma unroll
  for (int cc = 0; cc < 2; ++cc) {
    int c = lane + 32 * cc;
    float s = b1[c];
    for (int j = 0; j < Hh; ++j) s += hr[j] * w1[j * 64 + c];
    acc += tanhf(s) * w2[c];
  }
#pragma unroll
  for (int o = 16; o; o >>= 1) acc += __shfl_xor(acc, o, 32);
  if (lane == 0) a[i] = acc + b2[0];
}

// Per-graph softmax attention pooling + classifier head; one block per graph.
__global__ __launch_bounds__(256)
void att_cls(const float* __restrict__ h, const float* __restrict__ a,
             const float* __restrict__ w1, const float* __restrict__ b1,
             const float* __restrict__ w2, const float* __restrict__ b2,
             float* __restrict__ o_attw, float* __restrict__ o_log, int per_g)
{
  __shared__ float red[256];
  __shared__ float wsm[1024];
  __shared__ float gadd[Hh];
  __shared__ float hid[64];
  int g = blockIdx.x, t = threadIdx.x;
  const float* ab = a + (size_t)g * per_g;
  float mx = -3.4e38f;
  for (int r = t; r < per_g; r += 256) mx = fmaxf(mx, ab[r]);
  red[t] = mx; __syncthreads();
  for (int s = 128; s > 0; s >>= 1) { if (t < s) red[t] = fmaxf(red[t], red[t + s]); __syncthreads(); }
  float m = red[0]; __syncthreads();
  float sum = 0.f;
  for (int r = t; r < per_g; r += 256) { float e2 = __expf(ab[r] - m); wsm[r] = e2; sum += e2; }
  red[t] = sum; __syncthreads();
  for (int s = 128; s > 0; s >>= 1) { if (t < s) red[t] += red[t + s]; __syncthreads(); }
  float invz = 1.f / red[0]; __syncthreads();
  for (int r = t; r < per_g; r += 256) {
    float w = wsm[r] * invz;
    wsm[r] = w;
    o_attw[(size_t)g * per_g + r] = w;
  }
  __syncthreads();
  if (t < Hh) {
    const float* base = h + (size_t)g * per_g * Hh;
    float s2 = 0.f;
    for (int r = 0; r < per_g; ++r) s2 += base[(size_t)r * Hh + t] * wsm[r];
    gadd[t] = s2;
  }
  __syncthreads();
  if (t < 64) {
    float v = b1[t];
    for (int k = 0; k < Hh; ++k) v += gadd[k] * w1[k * 64 + t];
    hid[t] = fmaxf(v, 0.f);
  }
  __syncthreads();
  if (t < 3) {
    float v = b2[t];
    for (int k = 0; k < 64; ++k) v += hid[k] * w2[k * 3 + t];
    o_log[g * 3 + t] = v;
  }
}

// ---------------------------------------------------------------------------
extern "C" void kernel_launch(void* const* d_in, const int* in_sizes, int n_in,
                              void* d_out, int out_size, void* d_ws, size_t ws_size,
                              hipStream_t stream)
{
  // setup_inputs() order: x, edge_attr, gen_mask, params{in insertion order}, edge_index, batch
  const float* x         = (const float*)d_in[0];
  const float* edge_attr = (const float*)d_in[1];
  const float* gen_mask  = (const float*)d_in[2];
  const float* in_w      = (const float*)d_in[3];
  const float* in_b      = (const float*)d_in[4];
  const float* node_w    = (const float*)d_in[5];
  const float* node_b    = (const float*)d_in[6];
  const float* edge_w    = (const float*)d_in[7];
  const float* edge_b    = (const float*)d_in[8];
  const float* msg_w     = (const float*)d_in[9];
  const float* msg_b     = (const float*)d_in[10];
  const float* ln_g      = (const float*)d_in[11];
  const float* ln_b      = (const float*)d_in[12];
  const float* pf_w1     = (const float*)d_in[13];
  const float* pf_b1     = (const float*)d_in[14];
  const float* pf_w2     = (const float*)d_in[15];
  const float* pf_b2     = (const float*)d_in[16];
  const float* gen_w1    = (const float*)d_in[17];
  const float* gen_b1    = (const float*)d_in[18];
  const float* gen_w2    = (const float*)d_in[19];
  const float* gen_b2    = (const float*)d_in[20];
  const float* cost_w1   = (const float*)d_in[21];
  const float* cost_b1   = (const float*)d_in[22];
  const float* cost_w2   = (const float*)d_in[23];
  const float* cost_b2   = (const float*)d_in[24];
  const float* att_w1    = (const float*)d_in[25];
  const float* att_b1    = (const float*)d_in[26];
  const float* att_w2    = (const float*)d_in[27];
  const float* att_b2    = (const float*)d_in[28];
  const float* cls_w1    = (const float*)d_in[29];
  const float* cls_b1    = (const float*)d_in[30];
  const float* cls_w2    = (const float*)d_in[31];
  const float* cls_b2    = (const float*)d_in[32];
  const int*   edge_idx  = (const int*)d_in[33];
  // d_in[34] = batch; structure (i / 1000) exploited directly.

  // Workspace carve (~91 MB total).
  char* wp = (char*)d_ws;
  auto take = [&](size_t bytes) -> void* {
    void* p = (void*)wp;
    wp += (bytes + 255) & ~(size_t)255;
    return p;
  };
  unsigned short* hb    = (unsigned short*)take((size_t)Nn * Hh * 2);
  float*          h     = (float*)take((size_t)Nn * Hh * 4);
  float*          hn2   = (float*)take((size_t)Nn * Hh * 4);
  float*          agg   = (float*)take((size_t)Nn * Hh * 4);
  float*          Weff  = (float*)take((size_t)Ll * Hh * Hh * 4);
  unsigned int*   Bpack = (unsigned int*)take((size_t)5 * PACK_DW * 4);
  float*          W2eff = (float*)take((size_t)Ll * 2 * Hh * 4);
  float*          beff  = (float*)take((size_t)Ll * Hh * 4);
  float*          asc   = (float*)take((size_t)Nn * 4);

  // Output carve (return order: v_mag, sin, cos, gen, cost, logits, att_w, h)
  float* out    = (float*)d_out;
  float* o_vmag = out;
  float* o_sin  = out + Nn;
  float* o_cos  = out + 2 * Nn;
  float* o_gen  = out + 3 * Nn;
  float* o_cost = out + 5 * Nn;
  float* o_log  = out + 5 * Nn + Gg;
  float* o_attw = out + 5 * Nn + 4 * Gg;
  float* o_h    = out + 6 * Nn + 4 * Gg;

  // Precompute fused weights & pack WMMA-B fragments.
  compute_weff<<<(Ll * Hh * Hh + 255) / 256, 256, 0, stream>>>(node_w, msg_w, Weff);
  for (int l = 0; l < Ll; ++l)
    pack_b<<<PACK_DW / 256, 256, 0, stream>>>(Weff + (size_t)l * Hh * Hh,
                                              Bpack + (size_t)l * PACK_DW);
  pack_b<<<PACK_DW / 256, 256, 0, stream>>>(pf_w1,  Bpack + (size_t)3 * PACK_DW);
  pack_b<<<PACK_DW / 256, 256, 0, stream>>>(gen_w1, Bpack + (size_t)4 * PACK_DW);
  prep_edge_const<<<(Ll * Hh + 255) / 256, 256, 0, stream>>>(
      node_b, edge_w, edge_b, msg_w, msg_b, W2eff, beff);

  input_proj<<<(Nn * Hh) / 256, 256, 0, stream>>>(x, in_w, in_b, h, hb);

  const int gemm_blocks = (Nn / 16 + 7) / 8; // 3125 row tiles, 8 waves/block
  for (int l = 0; l < Ll; ++l) {
    gemm_bf16_wmma<<<gemm_blocks, 256, 0, stream>>>(
        hb, Bpack + (size_t)l * PACK_DW, beff + l * Hh, hn2, Nn, 0);
    zero_f4<<<(Nn * Hh / 4 + 255) / 256, 256, 0, stream>>>((float4*)agg, Nn * Hh / 4);
    edge_pass<<<(Ee * 32) / 256, 256, 0, stream>>>(
        hn2, edge_attr, edge_idx, W2eff + l * 2 * Hh, agg, Ee);
    layernorm_k<<<(Nn * 32) / 256, 256, 0, stream>>>(
        h, agg, ln_g + l * Hh, ln_b + l * Hh, (l == Ll - 1) ? o_h : h, hb, Nn);
  }

  // Heads: t1 = relu(h@pf_w1+b) in hn2, g1 = relu(h@gen_w1+b) in agg (both WMMA).
  gemm_bf16_wmma<<<gemm_blocks, 256, 0, stream>>>(
      hb, Bpack + (size_t)3 * PACK_DW, pf_b1, hn2, Nn, 1);
  gemm_bf16_wmma<<<gemm_blocks, 256, 0, stream>>>(
      hb, Bpack + (size_t)4 * PACK_DW, gen_b1, agg, Nn, 1);
  head_pf_gen<<<(Nn * 32) / 256, 256, 0, stream>>>(
      hn2, agg, pf_w2, pf_b2, gen_w2, gen_b2, gen_mask,
      o_vmag, o_sin, o_cos, o_gen, Nn);

  graph_cost<<<Gg, 128, 0, stream>>>(o_h, cost_w1, cost_b1, cost_w2, cost_b2, o_cost, Nn / Gg);
  att_score<<<(Nn * 32) / 256, 256, 0, stream>>>(o_h, att_w1, att_b1, att_w2, att_b2, asc, Nn);
  att_cls<<<Gg, 256, 0, stream>>>(o_h, asc, cls_w1, cls_b1, cls_w2, cls_b2, o_attw, o_log, Nn / Gg);

  (void)in_sizes; (void)n_in; (void)out_size; (void)ws_size;
}
